// FactorizedAttention_50938312130581
// MI455X (gfx1250) — matrix operand order repaired and checked by reading, not compiled, not run on hardware
//
#include <hip/hip_runtime.h>

// ---------------- Types ----------------
typedef _Float16 v16h __attribute__((ext_vector_type(16)));
typedef _Float16 h8   __attribute__((ext_vector_type(8)));
typedef _Float16 h4   __attribute__((ext_vector_type(4)));
typedef _Float16 h2   __attribute__((ext_vector_type(2)));
typedef float    v8f  __attribute__((ext_vector_type(8)));
typedef int      v4i  __attribute__((ext_vector_type(4)));

// ---------------- Problem constants ----------------
#define CTXN   4096
#define DMODEL 1024
#define NHEAD  16
#define DHEAD  64
#define BLEN   64
#define NBATCH 4
#define MTOK   (NBATCH * CTXN)   // 16384 tokens

// ---------------- CDNA5 async-to-LDS support (guarded) ----------------
#if defined(__has_builtin)
#  if __has_builtin(__builtin_amdgcn_global_load_async_to_lds_b128)
#    define HAVE_ASYNC_LDS 1
#  else
#    define HAVE_ASYNC_LDS 0
#  endif
#  if __has_builtin(__builtin_amdgcn_s_wait_asynccnt)
#    define HAVE_WAIT_ASYNC 1
#  else
#    define HAVE_WAIT_ASYNC 0
#  endif
#else
#  define HAVE_ASYNC_LDS 0
#  define HAVE_WAIT_ASYNC 0
#endif

// Builtin signature (from hipcc diagnostic): arg0 is "int4 __device__*"
// i.e. addrspace(1) v4i*, arg1 is the LDS-side pointer (addrspace(3)).
typedef __attribute__((address_space(1))) v4i as1_v4i;
typedef __attribute__((address_space(3))) v4i as3_v4i;

__device__ __forceinline__ void asyncWait0()
{
#if HAVE_ASYNC_LDS
#  if HAVE_WAIT_ASYNC
    __builtin_amdgcn_s_wait_asynccnt(0);
#  else
    asm volatile("s_wait_asynccnt 0x0" ::: "memory");
#  endif
#endif
}

// 16-byte global -> LDS copy: async on CDNA5, synchronous fallback otherwise.
__device__ __forceinline__ void copy16B(const _Float16* __restrict__ src,
                                        _Float16* __restrict__ dstLds)
{
#if HAVE_ASYNC_LDS
    __builtin_amdgcn_global_load_async_to_lds_b128((as1_v4i*)src, (as3_v4i*)dstLds, 0, 0);
#else
    *(h8*)dstLds = *(const h8*)src;
#endif
}

// ============================================================
// Kernel 0: convert + transpose the 4 weight matrices to f16.
//   Wq/Wk/Wv: BT[n=h*64+c][k=d] = W[d][h*64+c]
//   Wo:       BT[n=d][k=h*64+c] = Wo[h*64+c][d]
// (all four are plain 1024x1024 transposes)
// ============================================================
__global__ __launch_bounds__(256) void convert_weights(
    const float* __restrict__ Wq, const float* __restrict__ Wk,
    const float* __restrict__ Wv, const float* __restrict__ Wo,
    _Float16* __restrict__ wq, _Float16* __restrict__ wk,
    _Float16* __restrict__ wv, _Float16* __restrict__ wo)
{
    int idx = blockIdx.x * 256 + threadIdx.x;     // 0 .. 1024*1024-1
    int r = idx >> 10, c = idx & 1023;
    int src = c * 1024 + r;
    wq[idx] = (_Float16)Wq[src];
    wk[idx] = (_Float16)Wk[src];
    wv[idx] = (_Float16)Wv[src];
    wo[idx] = (_Float16)Wo[src];
}

// ============================================================
// WMMA GEMM: Y[M x 1024] = A[M x 1024] @ BT^T + bias*bscale
//   BT is [N=1024][K=1024] f16 (pre-transposed, so B-frags are contiguous).
//   A_F32:  A is f32 (converted to f16 on LDS stage-in), A2 unused.
//   !A_F32: A and A2 are f16, summed on stage-in (final proj: col+row).
//   OUT_F32: write f32 (final output), else f16 (qh/kh/vh).
// 256 threads = 8 waves; tile 128x128; K stage = 64 (2 WMMA k-steps).
// Double-buffered LDS, software pipelined:
//   A staged via registers (needs convert/add), B staged async-to-LDS.
// ============================================================
#define KSTG 64
#define AS2  72   // LDS row stride in halves (64 + 8 pad)

template<bool A_F32, bool OUT_F32>
__global__ __launch_bounds__(256) void gemm128(
    const void* __restrict__ Ap, const void* __restrict__ A2p,
    const _Float16* __restrict__ BT, const float* __restrict__ bias,
    float bscale, void* __restrict__ Yp)
{
    __shared__ _Float16 Ash[2][128 * AS2];
    __shared__ _Float16 Bsh[2][128 * AS2];

    const int t    = threadIdx.x;
    const int mblk = blockIdx.x, nblk = blockIdx.y;
    const int lane = t & 31, wave = t >> 5;
    const int wm = wave & 1, wn = wave >> 1;
    const int lr = lane & 15, hf = lane >> 4;

    v8f acc[4][2];
#pragma unroll
    for (int mt = 0; mt < 4; ++mt)
#pragma unroll
        for (int nt = 0; nt < 2; ++nt) acc[mt][nt] = (v8f)0.0f;

    // A prefetch registers
    float4 fA[8];
    h8 hA[4], hA2[4];

    auto issueA = [&](int kk) {
        if constexpr (A_F32) {
            const float* A = (const float*)Ap;
#pragma unroll
            for (int i = 0; i < 8; ++i) {
                int idx = t + i * 256;          // 2048 float4 slots (128x64 f32)
                int r = idx >> 4, c4 = idx & 15;
                fA[i] = *(const float4*)(A + (size_t)(mblk * 128 + r) * 1024 + kk + c4 * 4);
            }
        } else {
            const h8* A  = (const h8*)Ap;
            const h8* A2 = (const h8*)A2p;
#pragma unroll
            for (int i = 0; i < 4; ++i) {
                int idx = t + i * 256;          // 1024 h8 slots (128x64 f16)
                int r = idx >> 3, c8 = idx & 7;
                size_t off = (size_t)(mblk * 128 + r) * 128 + (kk >> 3) + c8;
                hA[i]  = A[off];
                hA2[i] = A2[off];
            }
        }
    };
    auto storeA = [&](int buf) {
        if constexpr (A_F32) {
#pragma unroll
            for (int i = 0; i < 8; ++i) {
                int idx = t + i * 256;
                int r = idx >> 4, c4 = idx & 15;
                h4 hv;
                hv.x = (_Float16)fA[i].x; hv.y = (_Float16)fA[i].y;
                hv.z = (_Float16)fA[i].z; hv.w = (_Float16)fA[i].w;
                *(h4*)&Ash[buf][r * AS2 + c4 * 4] = hv;
            }
        } else {
#pragma unroll
            for (int i = 0; i < 4; ++i) {
                int idx = t + i * 256;
                int r = idx >> 3, c8 = idx & 7;
                *(h8*)&Ash[buf][r * AS2 + c8 * 8] = hA[i] + hA2[i];
            }
        }
    };
    auto issueB = [&](int kk, int buf) {
#pragma unroll
        for (int i = 0; i < 4; ++i) {
            int idx = t + i * 256;              // 1024 b128 chunks (128x64 f16)
            int r = idx >> 3, c8 = idx & 7;
            copy16B(BT + (size_t)(nblk * 128 + r) * 1024 + kk + c8 * 8,
                    &Bsh[buf][r * AS2 + c8 * 8]);
        }
    };
    auto compute = [&](int buf) {
#pragma unroll
        for (int ks2 = 0; ks2 < 2; ++ks2) {
            v16h a[4], b[2];
#pragma unroll
            for (int mt = 0; mt < 4; ++mt) {
                const _Float16* ab = &Ash[buf][(wm * 64 + mt * 16 + lr) * AS2 + ks2 * 32 + hf * 8];
                h8* av = (h8*)&a[mt];
                av[0] = *(const h8*)ab;
                av[1] = *(const h8*)(ab + 16);
            }
#pragma unroll
            for (int nt = 0; nt < 2; ++nt) {
                const _Float16* bb = &Bsh[buf][(wn * 32 + nt * 16 + lr) * AS2 + ks2 * 32 + hf * 8];
                h8* bv = (h8*)&b[nt];
                bv[0] = *(const h8*)bb;
                bv[1] = *(const h8*)(bb + 16);
            }
#pragma unroll
            for (int mt = 0; mt < 4; ++mt)
#pragma unroll
                for (int nt = 0; nt < 2; ++nt)
                    acc[mt][nt] = __builtin_amdgcn_wmma_f32_16x16x32_f16(
                        false, a[mt], false, b[nt], (short)0, acc[mt][nt], false, false);
        }
    };

    // ---- software pipeline over 16 stages of K=64 ----
    issueA(0);
    issueB(0, 0);
    storeA(0);
    asyncWait0();
    __syncthreads();

    for (int s = 0; s < 16; ++s) {
        const int buf = s & 1, nxt = buf ^ 1;
        if (s < 15) {
            issueA((s + 1) * KSTG);
            issueB((s + 1) * KSTG, nxt);
        }
        compute(buf);
        if (s < 15) {
            storeA(nxt);
            asyncWait0();
        }
        __syncthreads();
    }

    // ---- epilogue: bias + store ----
#pragma unroll
    for (int mt = 0; mt < 4; ++mt)
#pragma unroll
        for (int nt = 0; nt < 2; ++nt) {
            int ng = nblk * 128 + wn * 32 + nt * 16 + lr;
            float bv = bias[ng] * bscale;
#pragma unroll
            for (int r = 0; r < 8; ++r) {
                int mg = mblk * 128 + wm * 64 + mt * 16 + hf * 8 + r;
                float v = acc[mt][nt][r] + bv;
                if (OUT_F32) ((float*)Yp)[(size_t)mg * 1024 + ng] = v;
                else ((_Float16*)Yp)[(size_t)mg * 1024 + ng] = (_Float16)v;
            }
        }
}

// ============================================================
// 64x64 causal attention over one (sequence, head).
// COLMODE: sequence = (n, o): rows are blocks, token = n*4096 + row*64 + o,
//          causal over blocks.
// !COLMODE (row mode): sequence = (n, b): Q tokens n*4096+b*64+i, K/V from
//          block b-1 (zeros for b==0), causal within block.
// 128 threads = 4 waves; wave w owns S rows [16w, 16w+16).
// K staged async-to-LDS (row-major = QK^T B operand);
// V staged manually transposed (PV B operand needs V^T).
// ============================================================
#define KS 72   // LDS row stride in halves

template<bool COLMODE>
__global__ __launch_bounds__(128) void attn64(
    const _Float16* __restrict__ qh, const _Float16* __restrict__ kh,
    const _Float16* __restrict__ vh, _Float16* __restrict__ outh)
{
    __shared__ _Float16 Ksh[64 * KS];    // K rows j, cols c  (= B for Q K^T)
    __shared__ _Float16 VTsh[64 * KS];   // V transposed: [c][j] (= B for P V)
    __shared__ _Float16 Psh[64 * KS];    // softmax probs [i][j]

    const int t = threadIdx.x;
    const int lane = t & 31, wave = t >> 5;
    const int lr = lane & 15, hf = lane >> 4;
    const int bx = blockIdx.x;           // 0..255
    const int h  = blockIdx.y;           // 0..15
    const int n  = bx >> 6;
    const int sub = bx & 63;             // o (col) or b (row)
    const size_t hoff = (size_t)h * 64;

    const bool kvalid = COLMODE || (sub > 0);

    // ---- stage K rows (async when possible; zero-fill row-mode block 0) ----
    if (kvalid) {
#pragma unroll
        for (int i = 0; i < 4; ++i) {
            int idx = t + i * 128;       // 512 b128 chunks (64x64 f16)
            int j = idx >> 3, c8 = idx & 7;
            size_t tok = COLMODE ? ((size_t)n * CTXN + (size_t)j * 64 + sub)
                                 : ((size_t)n * CTXN + (size_t)(sub - 1) * 64 + j);
            copy16B(kh + tok * 1024 + hoff + c8 * 8, &Ksh[j * KS + c8 * 8]);
        }
    } else {
#pragma unroll
        for (int i = 0; i < 4; ++i) {
            int idx = t + i * 128;
            int j = idx >> 3, c8 = idx & 7;
            *(h8*)&Ksh[j * KS + c8 * 8] = (h8)(_Float16)0.0f;
        }
    }
    // ---- stage V transposed ----
#pragma unroll
    for (int i = 0; i < 16; ++i) {
        int idx = t + i * 128;           // 2048 half2
        int j = idx >> 5, c2 = idx & 31;
        h2 vv = (h2)(_Float16)0.0f;
        if (kvalid) {
            size_t tok = COLMODE ? ((size_t)n * CTXN + (size_t)j * 64 + sub)
                                 : ((size_t)n * CTXN + (size_t)(sub - 1) * 64 + j);
            vv = *(const h2*)(vh + tok * 1024 + hoff + c2 * 2);
        }
        VTsh[(c2 * 2) * KS + j]     = vv.x;
        VTsh[(c2 * 2 + 1) * KS + j] = vv.y;
    }
    asyncWait0();
    __syncthreads();

    // ---- S = Q K^T (A frags straight from global qh) ----
    v8f s[4];
#pragma unroll
    for (int nt = 0; nt < 4; ++nt) s[nt] = (v8f)0.0f;
#pragma unroll
    for (int ks = 0; ks < 2; ++ks) {
        int m = wave * 16 + lr;
        size_t tok = COLMODE ? ((size_t)n * CTXN + (size_t)m * 64 + sub)
                             : ((size_t)n * CTXN + (size_t)sub * 64 + m);
        const _Float16* qb = qh + tok * 1024 + hoff + ks * 32 + hf * 8;
        v16h a; h8* av = (h8*)&a;
        av[0] = *(const h8*)qb;
        av[1] = *(const h8*)(qb + 16);
#pragma unroll
        for (int nt = 0; nt < 4; ++nt) {
            const _Float16* kb = &Ksh[(nt * 16 + lr) * KS + ks * 32 + hf * 8];
            v16h b; h8* bv = (h8*)&b;
            bv[0] = *(const h8*)kb;
            bv[1] = *(const h8*)(kb + 16);
            s[nt] = __builtin_amdgcn_wmma_f32_16x16x32_f16(
                false, a, false, b, (short)0, s[nt], false, false);
        }
    }

    // ---- mask + softmax.  C layout: row = wave*16 + hf*8 + r, key = nt*16 + lr.
    float sv[4][8];
#pragma unroll
    for (int nt = 0; nt < 4; ++nt)
#pragma unroll
        for (int r = 0; r < 8; ++r) {
            int qi = wave * 16 + hf * 8 + r;
            int kj = nt * 16 + lr;
            float x = s[nt][r] * 0.125f;          // 1/sqrt(64)
            if (kj > qi) x = -1e9f;               // causal (blocks or in-block)
            sv[nt][r] = x;
        }
    float m8[8], sum8[8];
#pragma unroll
    for (int r = 0; r < 8; ++r)
        m8[r] = fmaxf(fmaxf(sv[0][r], sv[1][r]), fmaxf(sv[2][r], sv[3][r]));
#pragma unroll
    for (int d = 1; d < 16; d <<= 1)              // stays within 16-lane halves
#pragma unroll
        for (int r = 0; r < 8; ++r) m8[r] = fmaxf(m8[r], __shfl_xor(m8[r], d, 32));
#pragma unroll
    for (int r = 0; r < 8; ++r) sum8[r] = 0.0f;
    float p[4][8];
#pragma unroll
    for (int nt = 0; nt < 4; ++nt)
#pragma unroll
        for (int r = 0; r < 8; ++r) {
            float e = __expf(sv[nt][r] - m8[r]);
            p[nt][r] = e;
            sum8[r] += e;
        }
#pragma unroll
    for (int d = 1; d < 16; d <<= 1)
#pragma unroll
        for (int r = 0; r < 8; ++r) sum8[r] += __shfl_xor(sum8[r], d, 32);
#pragma unroll
    for (int r = 0; r < 8; ++r) sum8[r] = 1.0f / sum8[r];
#pragma unroll
    for (int nt = 0; nt < 4; ++nt)
#pragma unroll
        for (int r = 0; r < 8; ++r)
            Psh[(wave * 16 + hf * 8 + r) * KS + nt * 16 + lr] =
                (_Float16)(p[nt][r] * sum8[r]);
    __syncthreads();

    // ---- O = P @ V ----
    v8f o[4];
#pragma unroll
    for (int ct = 0; ct < 4; ++ct) o[ct] = (v8f)0.0f;
#pragma unroll
    for (int ks = 0; ks < 2; ++ks) {
        const _Float16* pb = &Psh[(wave * 16 + lr) * KS + ks * 32 + hf * 8];
        v16h a; h8* av = (h8*)&a;
        av[0] = *(const h8*)pb;
        av[1] = *(const h8*)(pb + 16);
#pragma unroll
        for (int ct = 0; ct < 4; ++ct) {
            const _Float16* vb = &VTsh[(ct * 16 + lr) * KS + ks * 32 + hf * 8];
            v16h b; h8* bv = (h8*)&b;
            bv[0] = *(const h8*)vb;
            bv[1] = *(const h8*)(vb + 16);
            o[ct] = __builtin_amdgcn_wmma_f32_16x16x32_f16(
                false, a, false, b, (short)0, o[ct], false, false);
        }
    }
    // ---- store per-head output (f16) ----
#pragma unroll
    for (int ct = 0; ct < 4; ++ct)
#pragma unroll
        for (int r = 0; r < 8; ++r) {
            int qi = wave * 16 + hf * 8 + r;
            size_t tok = COLMODE ? ((size_t)n * CTXN + (size_t)qi * 64 + sub)
                                 : ((size_t)n * CTXN + (size_t)sub * 64 + qi);
            outh[tok * 1024 + hoff + ct * 16 + lr] = (_Float16)o[ct][r];
        }
}

// ============================================================
// Launcher
// ============================================================
extern "C" void kernel_launch(void* const* d_in, const int* in_sizes, int n_in,
                              void* d_out, int out_size, void* d_ws, size_t ws_size,
                              hipStream_t stream)
{
    const float* q  = (const float*)d_in[0];
    const float* k  = (const float*)d_in[1];
    const float* v  = (const float*)d_in[2];
    const float* Wq = (const float*)d_in[3];
    const float* bq = (const float*)d_in[4];
    const float* Wk = (const float*)d_in[5];
    const float* bk = (const float*)d_in[6];
    const float* Wv = (const float*)d_in[7];
    const float* bv = (const float*)d_in[8];
    const float* Wo = (const float*)d_in[9];
    const float* bo = (const float*)d_in[10];

    // workspace layout (f16): 5 token matrices (32MB) + 4 weight matrices (2MB)
    const size_t SZH = (size_t)MTOK * 1024;
    _Float16* w    = (_Float16*)d_ws;
    _Float16* qhb  = w;
    _Float16* khb  = w + SZH;
    _Float16* vhb  = w + 2 * SZH;
    _Float16* colh = w + 3 * SZH;
    _Float16* rowh = w + 4 * SZH;
    _Float16* wq16 = w + 5 * SZH;
    _Float16* wk16 = wq16 + (size_t)1024 * 1024;
    _Float16* wv16 = wk16 + (size_t)1024 * 1024;
    _Float16* wo16 = wv16 + (size_t)1024 * 1024;

    convert_weights<<<4096, 256, 0, stream>>>(Wq, Wk, Wv, Wo, wq16, wk16, wv16, wo16);

    dim3 ggrid(MTOK / 128, 1024 / 128);
    gemm128<true, false><<<ggrid, 256, 0, stream>>>(q, nullptr, wq16, bq, 1.0f, qhb);
    gemm128<true, false><<<ggrid, 256, 0, stream>>>(k, nullptr, wk16, bk, 1.0f, khb);
    gemm128<true, false><<<ggrid, 256, 0, stream>>>(v, nullptr, wv16, bv, 1.0f, vhb);

    dim3 agrid(NBATCH * BLEN, NHEAD);        // (256, 16)
    attn64<true ><<<agrid, 128, 0, stream>>>(qhb, khb, vhb, colh);
    attn64<false><<<agrid, 128, 0, stream>>>(qhb, khb, vhb, rowh);

    // out = (colh + rowh) @ Wo + 2*bo   (bias appears once per branch)
    gemm128<false, true><<<ggrid, 256, 0, stream>>>(colh, rowh, wo16, bo, 2.0f, d_out);
}
